// SpatialAttentionRefinement_2121713844430
// MI455X (gfx1250) — matrix operand order repaired and checked by reading, not compiled, not run on hardware
//
#include <hip/hip_runtime.h>
#include <hip/hip_bf16.h>

typedef __attribute__((ext_vector_type(16))) _Float16 v16h;
typedef __attribute__((ext_vector_type(8)))  _Float16 v8h;
typedef __attribute__((ext_vector_type(8)))  float    v8f;

#define BATCH 64
#define CF    2048   // feat channels
#define MID   256    // projected channels
#define NSP   512    // H*W spatial
#define CC    2048   // cam channels

// ---------------- fragment helpers (wave32 WMMA f16 layouts per CDNA5 ISA) ---

// A-matrix 16x32 f16: lane L -> row m=L&15, h=L>>4; elems 0..7 = K h*8+0..7,
// elems 8..15 = K 16+h*8+0..7. Source row-major [row][k], stride in halfs.
__device__ __forceinline__ v16h frag_a_f16(const _Float16* __restrict__ base,
                                           int stride, int lane) {
    int m = lane & 15, h = lane >> 4;
    const _Float16* p = base + (size_t)m * stride + h * 8;
    union { v16h v; v8h h2[2]; } u;
    u.h2[0] = *(const v8h*)(p);
    u.h2[1] = *(const v8h*)(p + 16);
    return u.v;
}

// Same A layout but source is f32 row-major; convert in registers.
__device__ __forceinline__ v16h frag_a_f32(const float* __restrict__ base,
                                           int stride, int lane) {
    int m = lane & 15, h = lane >> 4;
    const float* p = base + (size_t)m * stride + h * 8;
    union { v16h v; _Float16 e[16]; } u;
#pragma unroll
    for (int q = 0; q < 8; ++q) u.e[q]     = (_Float16)p[q];
#pragma unroll
    for (int q = 0; q < 8; ++q) u.e[8 + q] = (_Float16)p[16 + q];
    return u.v;
}

// B-matrix 32x16 f16: lane L -> col n=L&15, g=L>>4; elems 0..7 = K g*8+0..7, elems 8..15 = K 16+g*8+0..7.
// Source stored K-contiguous per column: [n][k], stride in halfs.
__device__ __forceinline__ v16h frag_b_f16(const _Float16* __restrict__ base,
                                           int stride, int lane) {
    int n = lane & 15, g = lane >> 4;
    const _Float16* p = base + (size_t)n * stride + g * 8;
    union { v16h v; v8h h2[2]; } u;
    u.h2[0] = *(const v8h*)(p);
    u.h2[1] = *(const v8h*)(p + 16);
    return u.v;
}

__device__ __forceinline__ v8f wmma16(v16h a, v16h b, v8f c) {
    return __builtin_amdgcn_wmma_f32_16x16x32_f16(false, a, false, b,
                                                  (short)0, c, false, false);
}

// ============================================================================
// Kernel 1: G[m,n] = sum_c W[m,c] * feat[b,c,n]   (M=256, N=512, K=2048)
// Output stored transposed: g[b][n][m] (f16, m contiguous) for kernel 2.
// Block tile 128(M) x 64(N), 8 waves -> wave tile 32x32, K-step 32 via LDS.
// grid: (N/64=8, M/128=2, B*2) ; z selects batch and W1-vs-W2.
// ============================================================================
__global__ void __launch_bounds__(256)
k_gemm_g(const float* __restrict__ feat, const float* __restrict__ W1,
         const float* __restrict__ W2, _Float16* __restrict__ g1,
         _Float16* __restrict__ g2) {
    __shared__ __align__(16) _Float16 shW[128 * 40]; // [mLocal][k], pad 40
    __shared__ __align__(16) _Float16 shF[64 * 40];  // [nLocal][k], pad 40

    const int tid  = threadIdx.x;
    const int lane = tid & 31;
    const int w    = tid >> 5;
    const int wm   = w & 3;        // 4 x 32 rows
    const int wn   = w >> 2;       // 2 x 32 cols

    const int nBase = blockIdx.x * 64;
    const int mBase = blockIdx.y * 128;
    const int b     = blockIdx.z >> 1;
    const int sel   = blockIdx.z & 1;

    const float* __restrict__ Wp = sel ? W2 : W1;
    const float* __restrict__ fb = feat + (size_t)b * CF * NSP;

    const v8f zero = {0.f, 0.f, 0.f, 0.f, 0.f, 0.f, 0.f, 0.f};
    v8f acc[2][2];
#pragma unroll
    for (int i = 0; i < 2; ++i)
#pragma unroll
        for (int j = 0; j < 2; ++j) acc[i][j] = zero;

    for (int kk = 0; kk < CF; kk += 32) {
        // --- load W tile 128x32 f32 -> f16 LDS [m][k] ---
        {
            int row = tid >> 1, half = tid & 1;
            const float* src = Wp + (size_t)(mBase + row) * CF + kk + half * 16;
            _Float16* dst = &shW[row * 40 + half * 16];
            union { v8h v; _Float16 e[8]; } u0, u1;
#pragma unroll
            for (int q = 0; q < 8; ++q) u0.e[q] = (_Float16)src[q];
#pragma unroll
            for (int q = 0; q < 8; ++q) u1.e[q] = (_Float16)src[8 + q];
            *(v8h*)(dst)     = u0.v;
            *(v8h*)(dst + 8) = u1.v;
        }
        // --- load feat tile 32(k) x 64(n) f32, transpose -> f16 LDS [n][k] ---
        {
            int nPair = (tid & 31) * 2;
#pragma unroll
            for (int kr = 0; kr < 4; ++kr) {
                int k = kr * 8 + (tid >> 5);
                float2 v = *(const float2*)(fb + (size_t)(kk + k) * NSP + nBase + nPair);
                shF[(nPair)     * 40 + k] = (_Float16)v.x;
                shF[(nPair + 1) * 40 + k] = (_Float16)v.y;
            }
        }
        __syncthreads();

#pragma unroll
        for (int a16 = 0; a16 < 2; ++a16) {
            v16h af = frag_a_f16(&shW[(wm * 32 + a16 * 16) * 40], 40, lane);
#pragma unroll
            for (int b16 = 0; b16 < 2; ++b16) {
                v16h bf = frag_b_f16(&shF[(wn * 32 + b16 * 16) * 40], 40, lane);
                acc[a16][b16] = wmma16(af, bf, acc[a16][b16]);
            }
        }
        __syncthreads();
    }

    // store transposed: g[b][n][m], D lane holds (m = r + (lane>>4)*8, n = lane&15)
    _Float16* g = (sel ? g2 : g1) + (size_t)b * NSP * MID;
#pragma unroll
    for (int a16 = 0; a16 < 2; ++a16)
#pragma unroll
        for (int b16 = 0; b16 < 2; ++b16) {
            int n  = nBase + wn * 32 + b16 * 16 + (lane & 15);
            int m0 = mBase + wm * 32 + a16 * 16 + (lane >> 4) * 8;
            union { v8h v; _Float16 e[8]; } u;
#pragma unroll
            for (int r = 0; r < 8; ++r) u.e[r] = (_Float16)acc[a16][b16][r];
            *(v8h*)(g + (size_t)n * MID + m0) = u.v;
        }
}

// ============================================================================
// Kernel 2: S[j,i] = sum_m g2[b][j][m]*g1[b][i][m]; softmax over i;
// write attnT[b][i][j] (f16, j contiguous) for kernel 3.
// Block: 32 J-rows x full 512 I, K=256. 8 waves -> wave tile 16(J)x128(I).
// grid: (NSP/32=16, B)
// ============================================================================
__global__ void __launch_bounds__(256)
k_attn(const _Float16* __restrict__ g1, const _Float16* __restrict__ g2,
       _Float16* __restrict__ attnT) {
    __shared__ float S[32 * 520];   // [jLocal][i], pad 520 vs 512
    __shared__ float redMax[256];
    __shared__ float redSum[256];
    __shared__ float invS[32];

    const int tid  = threadIdx.x;
    const int lane = tid & 31;
    const int w    = tid >> 5;
    const int jSub = (w & 1) * 16;
    const int iSub = (w >> 1) * 128;
    const int j0   = blockIdx.x * 32;
    const int b    = blockIdx.y;

    const _Float16* g1b = g1 + (size_t)b * NSP * MID;
    const _Float16* g2b = g2 + (size_t)b * NSP * MID;

    const v8f zero = {0.f, 0.f, 0.f, 0.f, 0.f, 0.f, 0.f, 0.f};
    v8f acc[8];
#pragma unroll
    for (int t = 0; t < 8; ++t) acc[t] = zero;

    for (int k0 = 0; k0 < MID; k0 += 32) {
        v16h af = frag_a_f16(g2b + (size_t)(j0 + jSub) * MID + k0, MID, lane);
#pragma unroll
        for (int t = 0; t < 8; ++t) {
            v16h bf = frag_b_f16(g1b + (size_t)(iSub + t * 16) * MID + k0, MID, lane);
            acc[t] = wmma16(af, bf, acc[t]);
        }
    }

    // spill S tile to LDS
#pragma unroll
    for (int t = 0; t < 8; ++t)
#pragma unroll
        for (int r = 0; r < 8; ++r) {
            int jl = jSub + r + (lane >> 4) * 8;
            int il = iSub + t * 16 + (lane & 15);
            S[jl * 520 + il] = acc[t][r];
        }
    __syncthreads();

    // softmax over i: 8 threads per row, 64 cols each
    const int row = tid >> 3, seg = tid & 7;
    float* Srow = &S[row * 520];
    float mx = -3.0e38f;
    for (int c = seg * 64; c < seg * 64 + 64; ++c) mx = fmaxf(mx, Srow[c]);
    redMax[tid] = mx;
    __syncthreads();
    float rm = redMax[row * 8];
#pragma unroll
    for (int q = 1; q < 8; ++q) rm = fmaxf(rm, redMax[row * 8 + q]);
    float sum = 0.f;
    for (int c = seg * 64; c < seg * 64 + 64; ++c) {
        float e = __expf(Srow[c] - rm);
        Srow[c] = e;
        sum += e;
    }
    redSum[tid] = sum;
    __syncthreads();
    if (seg == 0) {
        float tot = 0.f;
#pragma unroll
        for (int q = 0; q < 8; ++q) tot += redSum[row * 8 + q];
        invS[row] = 1.0f / tot;
    }
    __syncthreads();

    // write attnT[b][i][j0..j0+31] transposed, packed 16B stores
    _Float16* dstB = attnT + (size_t)b * NSP * NSP + j0;
#pragma unroll
    for (int rep = 0; rep < 2; ++rep) {
        int i = tid + rep * 256;
        _Float16* dst = dstB + (size_t)i * NSP;
#pragma unroll
        for (int q = 0; q < 4; ++q) {
            union { v8h v; _Float16 e[8]; } u;
#pragma unroll
            for (int jj = 0; jj < 8; ++jj) {
                int j = q * 8 + jj;
                u.e[jj] = (_Float16)(S[j * 520 + i] * invS[j]);
            }
            *(v8h*)(dst + q * 8) = u.v;
        }
    }
}

// ============================================================================
// Kernel 3: out[b,c,i] = alpha * sum_j cam[b,c,j] * attnT[b][i][j]
// Block tile 128(C) x 64(I), K=512. 8 waves -> wave tile 32x32.
// grid: (NSP/64=8, CC/128=16, B)
// ============================================================================
__global__ void __launch_bounds__(256)
k_out(const float* __restrict__ cam, const _Float16* __restrict__ attnT,
      const float* __restrict__ alphaPtr, float* __restrict__ out) {
    const int tid  = threadIdx.x;
    const int lane = tid & 31;
    const int w    = tid >> 5;
    const int cSub = (w & 3) * 32;
    const int iSub = (w >> 2) * 32;
    const int i0   = blockIdx.x * 64;
    const int c0   = blockIdx.y * 128;
    const int b    = blockIdx.z;

    const float alpha = alphaPtr[0];
    const float*    camb = cam   + (size_t)b * CC * NSP;
    const _Float16* atb  = attnT + (size_t)b * NSP * NSP;

    const v8f zero = {0.f, 0.f, 0.f, 0.f, 0.f, 0.f, 0.f, 0.f};
    v8f acc[2][2];
#pragma unroll
    for (int i = 0; i < 2; ++i)
#pragma unroll
        for (int j = 0; j < 2; ++j) acc[i][j] = zero;

    for (int k0 = 0; k0 < NSP; k0 += 32) {
#pragma unroll
        for (int a16 = 0; a16 < 2; ++a16) {
            v16h af = frag_a_f32(camb + (size_t)(c0 + cSub + a16 * 16) * NSP + k0,
                                 NSP, lane);
#pragma unroll
            for (int b16 = 0; b16 < 2; ++b16) {
                v16h bf = frag_b_f16(atb + (size_t)(i0 + iSub + b16 * 16) * NSP + k0,
                                     NSP, lane);
                acc[a16][b16] = wmma16(af, bf, acc[a16][b16]);
            }
        }
    }

    float* ob = out + (size_t)b * CC * NSP;
#pragma unroll
    for (int a16 = 0; a16 < 2; ++a16)
#pragma unroll
        for (int b16 = 0; b16 < 2; ++b16) {
            int i = i0 + iSub + b16 * 16 + (lane & 15);
            int c = c0 + cSub + a16 * 16 + (lane >> 4) * 8;
#pragma unroll
            for (int r = 0; r < 8; ++r)
                ob[(size_t)(c + r) * NSP + i] = alpha * acc[a16][b16][r];
        }
}

// ============================================================================
extern "C" void kernel_launch(void* const* d_in, const int* in_sizes, int n_in,
                              void* d_out, int out_size, void* d_ws, size_t ws_size,
                              hipStream_t stream) {
    const float* cam   = (const float*)d_in[0];
    const float* feat  = (const float*)d_in[1];
    const float* W1    = (const float*)d_in[2];
    const float* W2    = (const float*)d_in[3];
    const float* alpha = (const float*)d_in[4];
    float* out = (float*)d_out;

    _Float16* wsh   = (_Float16*)d_ws;
    const size_t GE = (size_t)BATCH * NSP * MID;      // 8.4M halfs each
    _Float16* g1    = wsh;
    _Float16* g2    = wsh + GE;
    _Float16* attnT = wsh + 2 * GE;                   // B*512*512 halfs

    k_gemm_g<<<dim3(NSP / 64, MID / 128, BATCH * 2), 256, 0, stream>>>(
        feat, W1, W2, g1, g2);
    k_attn<<<dim3(NSP / 32, BATCH), 256, 0, stream>>>(g1, g2, attnT);
    k_out<<<dim3(NSP / 64, CC / 128, BATCH), 256, 0, stream>>>(
        cam, attnT, alpha, out);
}
